// sheafBuilderDiag_62998580297939
// MI455X (gfx1250) — matrix operand order repaired and checked by reading, not compiled, not run on hardware
//
#include <hip/hip_runtime.h>
#include <hip/hip_bf16.h>
#include <stdint.h>

// ---------------------------------------------------------------------------
// SheafBuilderDiag for MI455X (gfx1250, wave32, WMMA).
//  Algebraic refactor: LayerNorm affine folded into W.
//    out[n] = rs*( h @ W' - mu*S[n] ) + c0[n],
//    W'[k][n] = scale[k]*W[k][n],  S[n] = sum_k W'[k][n],
//    c0[n]   = sum_k bias[k]*W[k][n] + b[n].
//  So the WMMA consumes raw gathered h in f16 (single pass: gather + stats +
//  pack + WMMA), and the mu/rs correction is a 2-op epilogue per output.
//  This kills the LDS staging tile and keeps VGPRs low for occupancy.
// ---------------------------------------------------------------------------

typedef __attribute__((ext_vector_type(16))) _Float16 v16h;
typedef __attribute__((ext_vector_type(8)))  float    v8f;

#define SB_D       6
#define SB_F       128
#define SB_NNODES  50000
#define SB_NEDGES  25000
#define SB_NNZ     1000000
#define SB_LN_EPS  1e-5f

#define WAVES_PER_BLOCK 4

// ---------------- Stage 1a: mean over the 6 stalk copies --------------------
__global__ void sheaf_mean6(const float* __restrict__ in,
                            float* __restrict__ out, int nrows) {
    int t = blockIdx.x * blockDim.x + threadIdx.x;
    int total = nrows * (SB_F / 4);
    if (t >= total) return;
    int nrow = t >> 5;          // F/4 == 32
    int f4   = t & 31;
    const float4* src = (const float4*)in;
    size_t base = (size_t)nrow * SB_D * (SB_F / 4) + f4;
    float4 a = src[base];
    #pragma unroll
    for (int k = 1; k < SB_D; ++k) {
        float4 v = src[base + (size_t)k * (SB_F / 4)];
        a.x += v.x; a.y += v.y; a.z += v.z; a.w += v.w;
    }
    const float inv = 1.0f / (float)SB_D;
    a.x *= inv; a.y *= inv; a.z *= inv; a.w *= inv;
    ((float4*)out)[t] = a;
}

// ---------------- Stage 1b: W' = scale .* W, padded 256x6 -> 256x16 f16 -----
// wpad[n*256 + k] = f16(scale[k] * W[k][n]) for n<6 else 0 — B-fragment order.
__global__ void sheaf_build_wpad(const float* __restrict__ W,
                                 const float* __restrict__ ln_scale,
                                 _Float16* __restrict__ wpad) {
    int k = threadIdx.x;          // 0..255
    float s = ln_scale[k];
    #pragma unroll
    for (int n = 0; n < 16; ++n) {
        float v = (n < SB_D) ? (s * W[k * SB_D + n]) : 0.0f;
        wpad[n * 256 + k] = (_Float16)v;
    }
}

// ---------------- Stage 1c: epilogue constants S[n], c0[n] ------------------
__global__ void sheaf_build_consts(const float* __restrict__ W,
                                   const float* __restrict__ ln_scale,
                                   const float* __restrict__ ln_bias,
                                   const float* __restrict__ bvec,
                                   float* __restrict__ Sn,
                                   float* __restrict__ C0n) {
    int n = threadIdx.x;          // 0..15
    if (n >= 16) return;
    float s = 0.0f, c0 = 0.0f;
    if (n < SB_D) {
        for (int k = 0; k < 2 * SB_F; ++k) {
            float w = W[k * SB_D + n];
            s  += ln_scale[k] * w;
            c0 += ln_bias[k] * w;
        }
        c0 += bvec[n];
    }
    Sn[n]  = s;
    C0n[n] = c0;
}

// ---------------- Stage 2: gather + stats + f16 WMMA + LN-epilogue ----------
__global__ void __launch_bounds__(WAVES_PER_BLOCK * 32)
sheaf_main(const float* __restrict__ xm, const float* __restrict__ em,
           const int* __restrict__ hye,
           const _Float16* __restrict__ wpad,
           const float* __restrict__ Sn, const float* __restrict__ C0n,
           int* __restrict__ idx_out, float* __restrict__ attr_out) {
    __shared__ __align__(16) _Float16 s_w[16 * 256];   // 8 KB, whole block

    {
        const uint32_t* ws = (const uint32_t*)wpad;
        uint32_t* wd = (uint32_t*)s_w;
        for (int i = threadIdx.x; i < 2048; i += WAVES_PER_BLOCK * 32)
            wd[i] = ws[i];
    }
    __syncthreads();

    const int lane = threadIdx.x & 31;
    const int wave = threadIdx.x >> 5;
    const int hlf  = lane >> 4;     // selects K sub-stripe of the fragment
    const int m    = lane & 15;     // A-row / B,C-column role
    const int tile = blockIdx.x * WAVES_PER_BLOCK + wave;
    const int base = tile * 16;     // first nnz row of this tile
    const int i    = base + m;      // nnz row handled by lanes m and m+16

    const int row = hye[i];
    const int col = hye[SB_NNZ + i];
    const float* __restrict__ xrow = xm + (size_t)row * SB_F;
    const float* __restrict__ erow = em + (size_t)col * SB_F;

    // Single fused pass: gather -> stats (f32) -> f16 pack -> WMMA.
    float sum = 0.0f, sq = 0.0f;
    v8f c = {};
    #pragma unroll
    for (int kc = 0; kc < 8; ++kc) {
        v16h a;
        #pragma unroll
        for (int t = 0; t < 2; ++t) {
            const int kb = 32 * kc + 16 * hlf + 8 * t;  // multiple of 8
            const float* p = (kb < SB_F) ? (xrow + kb) : (erow + (kb - SB_F));
            float4 v0 = *(const float4*)p;
            float4 v1 = *(const float4*)(p + 4);
            sum += v0.x + v0.y + v0.z + v0.w + v1.x + v1.y + v1.z + v1.w;
            sq  += v0.x * v0.x + v0.y * v0.y + v0.z * v0.z + v0.w * v0.w
                 + v1.x * v1.x + v1.y * v1.y + v1.z * v1.z + v1.w * v1.w;
            a[8 * t + 0] = (_Float16)v0.x;
            a[8 * t + 1] = (_Float16)v0.y;
            a[8 * t + 2] = (_Float16)v0.z;
            a[8 * t + 3] = (_Float16)v0.w;
            a[8 * t + 4] = (_Float16)v1.x;
            a[8 * t + 5] = (_Float16)v1.y;
            a[8 * t + 6] = (_Float16)v1.z;
            a[8 * t + 7] = (_Float16)v1.w;
        }
        v16h bfrag = *(const v16h*)(s_w + m * 256 + 32 * kc + 16 * hlf);
        c = __builtin_amdgcn_wmma_f32_16x16x32_f16(
                false, a, false, bfrag, (short)0, c, false, false);
    }

    // LayerNorm stats for row m live split across the lane pair (l, l^16).
    sum += __shfl_xor(sum, 16, 32);
    sq  += __shfl_xor(sq, 16, 32);
    const float mu = sum * (1.0f / 256.0f);
    const float rs = rsqrtf(sq * (1.0f / 256.0f) - mu * mu + SB_LN_EPS);

    // Epilogue: out = rs*(dot - mu*S[n]) + c0[n], then sigmoid.
    // C/D layout: c[v] = D[M = v + 8*hlf][N = m].
    if (m < SB_D) {
        const float Sm  = Sn[m];
        const float C0m = C0n[m];
        #pragma unroll
        for (int v = 0; v < 8; ++v) {
            const int M = v + 8 * hlf;
            float val = rs * (c[v] - mu * Sm) + C0m;
            val = 1.0f / (1.0f + __expf(-val));
            attr_out[(size_t)(base + M) * SB_D + m] = val;
        }
    }

    // Expanded sparse indices: out[part][i*6 + k] = 6*idx + k.
    const int src = hlf ? col : row;
    int* ip = idx_out + (size_t)hlf * (size_t)SB_D * SB_NNZ + (size_t)i * SB_D;
    #pragma unroll
    for (int k = 0; k < SB_D; ++k)
        ip[k] = SB_D * src + k;
}

// ---------------------------------------------------------------------------
extern "C" void kernel_launch(void* const* d_in, const int* in_sizes, int n_in,
                              void* d_out, int out_size, void* d_ws, size_t ws_size,
                              hipStream_t stream) {
    const float* x        = (const float*)d_in[0];   // [300000, 128]
    const float* e        = (const float*)d_in[1];   // [150000, 128]
    const int*   hye      = (const int*)d_in[2];     // [2, 1000000]
    const float* ln_scale = (const float*)d_in[3];   // [256]
    const float* ln_bias  = (const float*)d_in[4];   // [256]
    const float* W        = (const float*)d_in[5];   // [256, 6]
    const float* b        = (const float*)d_in[6];   // [6]

    // Workspace: xm (6.4M f32) | em (3.2M f32) | wpad (4096 f16) | S,c0 (32 f32)
    float*    xm   = (float*)d_ws;
    float*    em   = xm + (size_t)SB_NNODES * SB_F;
    _Float16* wpad = (_Float16*)(em + (size_t)SB_NEDGES * SB_F);
    float*    Sn   = (float*)(wpad + 16 * 256);
    float*    C0n  = Sn + 16;

    // Output: [2*6M int32 indices | 6M f32 attributes], 4 bytes each.
    int*   idx_out  = (int*)d_out;
    float* attr_out = (float*)d_out + (size_t)2 * SB_D * SB_NNZ;

    {
        int total = SB_NNODES * (SB_F / 4);
        sheaf_mean6<<<(total + 255) / 256, 256, 0, stream>>>(x, xm, SB_NNODES);
    }
    {
        int total = SB_NEDGES * (SB_F / 4);
        sheaf_mean6<<<(total + 255) / 256, 256, 0, stream>>>(e, em, SB_NEDGES);
    }
    sheaf_build_wpad<<<1, 256, 0, stream>>>(W, ln_scale, wpad);
    sheaf_build_consts<<<1, 16, 0, stream>>>(W, ln_scale, ln_bias, b, Sn, C0n);

    // 1M nnz / 16 rows per wave / 4 waves per block = 15625 blocks (exact).
    sheaf_main<<<SB_NNZ / 16 / WAVES_PER_BLOCK, WAVES_PER_BLOCK * 32, 0, stream>>>(
        xm, em, hye, wpad, Sn, C0n, idx_out, attr_out);
}
